// HeterogeneousGNN_75118978007203
// MI455X (gfx1250) — matrix operand order repaired and checked by reading, not compiled, not run on hardware
//
#include <hip/hip_runtime.h>
#include <math.h>

// ---------------- problem constants ----------------
#define NNODE 50000
#define NEDGE 150000
#define NNET  4
#define DIM   256
#define INF_  768
#define BB    8
#define SEQ   128
#define NL    3
#define NC    14
#define HH    128
#define ALPHA 0.1f

// ---------------- WMMA types ----------------
typedef __attribute__((ext_vector_type(16))) __bf16 v16bf;
typedef __attribute__((ext_vector_type(8)))  float  v8f;

union FragBF { v16bf bf; unsigned u[8]; };

static __device__ __forceinline__ unsigned short f2bf_bits(float f) {
  unsigned u = __float_as_uint(f);
  unsigned r = u + 0x7FFFu + ((u >> 16) & 1u);   // round-to-nearest-even
  return (unsigned short)(r >> 16);
}
static __device__ __forceinline__ unsigned pack2bf(float a, float b) {
  return (unsigned)f2bf_bits(a) | ((unsigned)f2bf_bits(b) << 16);
}

// load 16 consecutive elements as 8 bf16-pair dwords (vectorized b128 loads)
static __device__ __forceinline__ void load16_bf(const float* __restrict__ p, unsigned out[8]) {
  const float4* p4 = (const float4*)p;
#pragma unroll
  for (int i = 0; i < 4; ++i) {
    float4 v = p4[i];
    out[2 * i + 0] = pack2bf(v.x, v.y);
    out[2 * i + 1] = pack2bf(v.z, v.w);
  }
}
static __device__ __forceinline__ void load16_bf(const unsigned short* __restrict__ p, unsigned out[8]) {
  const uint4* p4 = (const uint4*)p;
#pragma unroll
  for (int i = 0; i < 2; ++i) {
    uint4 v = p4[i];
    out[4 * i + 0] = v.x; out[4 * i + 1] = v.y;
    out[4 * i + 2] = v.z; out[4 * i + 3] = v.w;
  }
}

// ---------------- tiled WMMA GEMM: C[M,N] = act(A[M,K] @ W[N,K]^T + bias) ----
// A: fp32 or bf16-bits (ushort). W/bias: fp32 (converted to bf16 while staging).
// Requires: K % 32 == 0, N % 64 == 0 (true for all launches here).
// Double-buffered LDS: one barrier per K-step; next tile's global loads overlap
// with current tile's WMMAs.
#define BM 128
#define BN 64
#define BK 32

template <typename AT>
__global__ __launch_bounds__(256)
void gemm_wmma_bf16(const AT* __restrict__ A, const float* __restrict__ W,
                    const float* __restrict__ bias, float* __restrict__ C,
                    int M, int N, int K, int act)
{
  // bf16 pairs, dword-addressed; [2] = double buffer
  __shared__ __align__(16) unsigned As32[2][BM * (BK / 2)];
  __shared__ __align__(16) unsigned Bs32[2][BN * (BK / 2)];

  const int tid  = threadIdx.x;
  const int wid  = tid >> 5;
  const int lane = tid & 31;
  const int half = lane >> 4;     // 0/1
  const int idx  = lane & 15;     // 0..15
  const int waveM = wid >> 1;     // 0..3
  const int waveN = wid & 1;      // 0..1
  const int m0 = blockIdx.x * BM;
  const int n0 = blockIdx.y * BN;

  // ---- per-thread staging slots (no divergent guards: clamp row) ----
  // A: thread -> (row = tid>>1, 16-elem K chunk = (tid&1)*16)
  const int arow = tid >> 1;
  const int ak   = (tid & 1) * 16;
  const int agr  = min(m0 + arow, M - 1);        // clamped; garbage rows never stored
  const AT* aptr = A + (size_t)agr * K + ak;
  // B: thread -> (row = tid>>2, 8-elem K chunk = (tid&3)*8)
  const int brow = tid >> 2;
  const int bk   = (tid & 3) * 8;
  const int bgr  = min(n0 + brow, N - 1);
  const float* bptr = W + (size_t)bgr * K + bk;

  v8f acc[2][2] = {};

#define STAGE(K0, ABUF, BBUF)                                                   \
  do {                                                                          \
    unsigned t_[8];                                                             \
    load16_bf(aptr + (K0), t_);                                                 \
    uint4* d_ = (uint4*)&(ABUF)[arow * 16 + (ak >> 1)];                         \
    d_[0] = make_uint4(t_[0], t_[1], t_[2], t_[3]);                             \
    d_[1] = make_uint4(t_[4], t_[5], t_[6], t_[7]);                             \
    const float4* p4_ = (const float4*)(bptr + (K0));                           \
    float4 v0_ = p4_[0], v1_ = p4_[1];                                          \
    *(uint4*)&(BBUF)[brow * 16 + (bk >> 1)] =                                   \
        make_uint4(pack2bf(v0_.x, v0_.y), pack2bf(v0_.z, v0_.w),                \
                   pack2bf(v1_.x, v1_.y), pack2bf(v1_.z, v1_.w));               \
  } while (0)

  // prologue: stage tile 0 into buffer 0
  STAGE(0, As32[0], Bs32[0]);

  int buf = 0;
  for (int k0 = 0; k0 < K; k0 += BK) {
    __syncthreads();   // tile in As32[buf]/Bs32[buf] is ready; prev reads drained
    if (k0 + BK < K) {
      STAGE(k0 + BK, As32[buf ^ 1], Bs32[buf ^ 1]);
      if (k0 + 2 * BK < K) __builtin_prefetch(aptr + k0 + 2 * BK, 0, 1);
    }

    const unsigned* As = As32[buf];
    const unsigned* Bs = Bs32[buf];

    // ---- fragments per CDNA5 16-bit A(16x32) / B(32x16) layouts ----
    FragBF fa[2], fb[2];
#pragma unroll
    for (int i = 0; i < 2; ++i) {
      int mr = waveM * 32 + i * 16 + idx;
#pragma unroll
      for (int v = 0; v < 8; ++v) {
        // A: K = 2*(v&3) + 8*half + 16*(v>>2)  -> dword (v&3)+4*half+8*(v>>2)
        fa[i].u[v] = As[mr * 16 + (v & 3) + 4 * half + 8 * (v >> 2)];
      }
    }
#pragma unroll
    for (int j = 0; j < 2; ++j) {
      int nb = waveN * 32 + j * 16 + idx;
#pragma unroll
      for (int v = 0; v < 8; ++v) {
        // B: K = 2*(v&3) + 8*half + 16*(v>>2)  -> dword (v&3)+4*half+8*(v>>2)
        fb[j].u[v] = Bs[nb * 16 + (v & 3) + 4 * half + 8 * (v >> 2)];
      }
    }
#pragma unroll
    for (int i = 0; i < 2; ++i)
#pragma unroll
      for (int j = 0; j < 2; ++j)
        acc[i][j] = __builtin_amdgcn_wmma_f32_16x16x32_bf16(
            false, fa[i].bf, false, fb[j].bf, (short)0, acc[i][j], false, false);

    buf ^= 1;
  }
#undef STAGE

  // ---- epilogue: bias + activation + store (C/D layout: M=v+8*half, N=idx) ----
#pragma unroll
  for (int j = 0; j < 2; ++j) {
    int col = n0 + waveN * 32 + j * 16 + idx;
    float bv = (bias != nullptr) ? bias[col] : 0.f;
#pragma unroll
    for (int i = 0; i < 2; ++i) {
#pragma unroll
      for (int v = 0; v < 8; ++v) {
        int row = m0 + waveM * 32 + i * 16 + v + 8 * half;
        if (row < M) {
          float o = acc[i][j][v] + bv;
          if (act) o = fmaxf(o, 0.f);
          C[(size_t)row * N + col] = o;
        }
      }
    }
  }
}

// ---------------- elementwise helpers ----------------
__global__ void fillf_kernel(float* __restrict__ p, size_t n, float v) {
  size_t i = (size_t)blockIdx.x * blockDim.x + threadIdx.x;
  size_t st = (size_t)gridDim.x * blockDim.x;
  for (; i < n; i += st) p[i] = v;
}

// n must be a multiple of 4 (12.8M here)
__global__ void f32_to_bf16_kernel(const float4* __restrict__ in, uint2* __restrict__ out, size_t n4) {
  size_t i = (size_t)blockIdx.x * blockDim.x + threadIdx.x;
  size_t st = (size_t)gridDim.x * blockDim.x;
  for (; i < n4; i += st) {
    float4 v = in[i];
    out[i] = make_uint2(pack2bf(v.x, v.y), pack2bf(v.z, v.w));
  }
}

__global__ void softmax4_kernel(const float* __restrict__ imp, float* __restrict__ ew) {
  if (threadIdx.x == 0 && blockIdx.x == 0) {
    float m = imp[0];
    for (int i = 1; i < NNET; ++i) m = fmaxf(m, imp[i]);
    float e[NNET], s = 0.f;
    for (int i = 0; i < NNET; ++i) { e[i] = expf(imp[i] - m); s += e[i]; }
    for (int i = 0; i < NNET; ++i) ew[i] = e[i] / s;
  }
}

// ---------------- edge scatter-add: agg[dst] += ew * msg[src] ----------------
__global__ __launch_bounds__(DIM)
void scatter_edges_kernel(const float* __restrict__ msg, const int* __restrict__ src,
                          const int* __restrict__ dst, const float* __restrict__ ew,
                          float* __restrict__ agg)
{
  int e = blockIdx.x;
  int c = threadIdx.x;
  int s = src[e];
  int d = dst[e];
  float w = ew[0];
  atomicAdd(&agg[(size_t)d * DIM + c], w * msg[(size_t)s * DIM + c]);
}

// ---------------- hout = (1-a)*agg + hs + a*h ; LN ; h_new = LN(..)+h -------
__global__ __launch_bounds__(DIM)
void combine_ln_kernel(const float* __restrict__ agg, const float* __restrict__ hs,
                       const float* __restrict__ h, const float* __restrict__ gamma,
                       const float* __restrict__ beta, float* __restrict__ hnew)
{
  __shared__ float red[DIM];
  int row = blockIdx.x, i = threadIdx.x;
  size_t base = (size_t)row * DIM;
  float hv = h[base + i];
  float hout = (1.f - ALPHA) * agg[base + i] + hs[base + i] + ALPHA * hv;

  red[i] = hout;
  __syncthreads();
  for (int s = DIM / 2; s > 0; s >>= 1) { if (i < s) red[i] += red[i + s]; __syncthreads(); }
  float mean = red[0] * (1.f / DIM);
  __syncthreads();
  float d = hout - mean;
  red[i] = d * d;
  __syncthreads();
  for (int s = DIM / 2; s > 0; s >>= 1) { if (i < s) red[i] += red[i + s]; __syncthreads(); }
  float var = red[0] * (1.f / DIM);
  hnew[base + i] = d * rsqrtf(var + 1e-5f) * gamma[i] + beta[i] + hv;
}

// ---------------- batch mean pooling ----------------
__global__ __launch_bounds__(DIM)
void pool_sum_kernel(const float* __restrict__ h, const int* __restrict__ batch,
                     float* __restrict__ sums, float* __restrict__ counts)
{
  int n = blockIdx.x, c = threadIdx.x;
  int b = batch[n];
  atomicAdd(&sums[(size_t)b * DIM + c], h[(size_t)n * DIM + c]);
  if (c == 0) atomicAdd(&counts[b], 1.f);
}

__global__ void pool_div_kernel(const float* __restrict__ sums, const float* __restrict__ counts,
                                float* __restrict__ gf)
{
  int b = blockIdx.x, c = threadIdx.x;
  gf[b * DIM + c] = sums[b * DIM + c] / fmaxf(counts[b], 1.f);
}

// ---------------- one BiLSTM layer: grid = 2 (dir), block = 1024 -------------
__global__ __launch_bounds__(1024)
void lstm_layer_kernel(const float* __restrict__ in,   // [B*SEQ, 256]
                       const float* __restrict__ Wih,  // [2, 512, 256] (dir-major)
                       const float* __restrict__ Whh,  // [2, 512, 128]
                       const float* __restrict__ bih,  // [2, 512]
                       const float* __restrict__ bhh,  // [2, 512]
                       float* __restrict__ out)        // [B*SEQ, 256] (dir halves)
{
  int dir = blockIdx.x;
  const float* Wih_d = Wih + (size_t)dir * 4 * HH * (2 * HH);
  const float* Whh_d = Whh + (size_t)dir * 4 * HH * HH;
  const float* bih_d = bih + dir * 4 * HH;
  const float* bhh_d = bhh + dir * 4 * HH;

  int tid = threadIdx.x;
  int b = tid >> 7;        // 0..7
  int j = tid & 127;       // 0..127

  __shared__ __align__(16) float hsm[BB][HH];
  __shared__ __align__(16) float xts[BB][2 * HH];

  float cst = 0.f;
  hsm[b][j] = 0.f;
  __syncthreads();

  for (int t = 0; t < SEQ; ++t) {
    int tt = dir ? (SEQ - 1 - t) : t;
    // stage x_t for all batches: 8*256 floats, 2 per thread
#pragma unroll
    for (int i = 0; i < 2; ++i) {
      int e = tid + i * 1024;
      int bb = e >> 8, k = e & 255;
      xts[bb][k] = in[((size_t)bb * SEQ + tt) * (2 * HH) + k];
    }
    __syncthreads();

    float g[4];
#pragma unroll
    for (int gg = 0; gg < 4; ++gg) {
      int row = gg * HH + j;
      float a = bih_d[row] + bhh_d[row];
      const float4* wr = (const float4*)&Wih_d[(size_t)row * (2 * HH)];
      const float4* xv = (const float4*)xts[b];
      for (int k = 0; k < (2 * HH) / 4; ++k) {
        float4 w = wr[k], xx = xv[k];
        a += w.x * xx.x + w.y * xx.y + w.z * xx.z + w.w * xx.w;
      }
      const float4* hr = (const float4*)&Whh_d[(size_t)row * HH];
      const float4* hv4 = (const float4*)hsm[b];
      for (int k = 0; k < HH / 4; ++k) {
        float4 w = hr[k], hh = hv4[k];
        a += w.x * hh.x + w.y * hh.y + w.z * hh.z + w.w * hh.w;
      }
      g[gg] = a;
    }
    __syncthreads();   // all reads of hsm done

    float ig = 1.f / (1.f + expf(-g[0]));
    float fg = 1.f / (1.f + expf(-g[1]));
    float cg = tanhf(g[2]);
    float og = 1.f / (1.f + expf(-g[3]));
    cst = fg * cst + ig * cg;
    float hv = og * tanhf(cst);
    hsm[b][j] = hv;
    out[((size_t)b * SEQ + tt) * (2 * HH) + dir * HH + j] = hv;
    __syncthreads();
  }
}

// ---------------- final MLP head (grid = B, block = 256) ---------------------
__global__ __launch_bounds__(DIM)
void head_kernel(const float* __restrict__ gf, const float* __restrict__ seq,
                 const float* __restrict__ fW1, const float* __restrict__ fb1,
                 const float* __restrict__ fW2, const float* __restrict__ fb2,
                 const float* __restrict__ Wc,  const float* __restrict__ bc,
                 float* __restrict__ out)
{
  int b = blockIdx.x, tid = threadIdx.x;
  __shared__ __align__(16) float comb[2 * DIM];
  __shared__ __align__(16) float f1[DIM];
  __shared__ __align__(16) float f2[DIM / 2];

  comb[tid]       = gf[b * DIM + tid];
  comb[DIM + tid] = seq[((size_t)b * SEQ + (SEQ - 1)) * DIM + tid];
  __syncthreads();

  {
    float a = fb1[tid];
    const float4* w1 = (const float4*)&fW1[(size_t)tid * (2 * DIM)];
    const float4* cv = (const float4*)comb;
    for (int k = 0; k < (2 * DIM) / 4; ++k) {
      float4 w = w1[k], c = cv[k];
      a += w.x * c.x + w.y * c.y + w.z * c.z + w.w * c.w;
    }
    f1[tid] = fmaxf(a, 0.f);
  }
  __syncthreads();

  if (tid < DIM / 2) {
    float a2 = fb2[tid];
    const float4* w2 = (const float4*)&fW2[(size_t)tid * DIM];
    const float4* fv = (const float4*)f1;
    for (int k = 0; k < DIM / 4; ++k) {
      float4 w = w2[k], f = fv[k];
      a2 += w.x * f.x + w.y * f.y + w.z * f.z + w.w * f.w;
    }
    f2[tid] = fmaxf(a2, 0.f);
  }
  __syncthreads();

  if (tid < NC) {
    float a3 = bc[tid];
    const float* wc = &Wc[(size_t)tid * (DIM / 2)];
    for (int k = 0; k < DIM / 2; ++k) a3 += wc[k] * f2[k];
    out[b * NC + tid] = a3;
  }
}

// =====================================================================
extern "C" void kernel_launch(void* const* d_in, const int* in_sizes, int n_in,
                              void* d_out, int out_size, void* d_ws, size_t ws_size,
                              hipStream_t stream) {
  (void)in_sizes; (void)n_in; (void)out_size; (void)ws_size;

  const float* x      = (const float*)d_in[0];
  const int*   eidx   = (const int*)  d_in[1];
  const int*   batch  = (const int*)  d_in[2];
  const float* seqf   = (const float*)d_in[3];
  const float* Wp     = (const float*)d_in[4];
  const float* bp     = (const float*)d_in[5];
  const float* edgeW  = (const float*)d_in[6];
  const float* selfW  = (const float*)d_in[7];
  const float* selfb  = (const float*)d_in[8];
  const float* gamma  = (const float*)d_in[9];
  const float* beta   = (const float*)d_in[10];
  const float* imp    = (const float*)d_in[11];
  const float* Ws     = (const float*)d_in[12];
  const float* bs     = (const float*)d_in[13];
  const float* Wih    = (const float*)d_in[14];
  const float* Whh    = (const float*)d_in[15];
  const float* bih    = (const float*)d_in[16];
  const float* bhh    = (const float*)d_in[17];
  const float* fW1    = (const float*)d_in[18];
  const float* fb1    = (const float*)d_in[19];
  const float* fW2    = (const float*)d_in[20];
  const float* fb2    = (const float*)d_in[21];
  const float* Wc     = (const float*)d_in[22];
  const float* bc     = (const float*)d_in[23];
  float* out = (float*)d_out;

  // ---- workspace layout ----
  char* base = (char*)d_ws;
  size_t off = 0;
  auto alloc = [&](size_t bytes) -> void* {
    void* r = base + off;
    off = (off + bytes + 255) & ~(size_t)255;
    return r;
  };
  const size_t HS = (size_t)NNODE * DIM * sizeof(float);   // 51.2 MB
  float*          h0    = (float*)alloc(HS);
  float*          h1    = (float*)alloc(HS);
  float*          hs    = (float*)alloc(HS);
  float*          msg   = (float*)alloc(HS);
  float*          agg   = (float*)alloc(HS);
  unsigned short* hbf   = (unsigned short*)alloc((size_t)NNODE * DIM * sizeof(unsigned short));
  float*          ew    = (float*)alloc(NNET * sizeof(float));
  float*          sums  = (float*)alloc(BB * DIM * sizeof(float));
  float*          cnts  = (float*)alloc(BB * sizeof(float));
  float*          gf    = (float*)alloc(BB * DIM * sizeof(float));
  float*          s0    = (float*)alloc((size_t)BB * SEQ * DIM * sizeof(float));
  float*          s1    = (float*)alloc((size_t)BB * SEQ * DIM * sizeof(float));
  float*          s2    = (float*)alloc((size_t)BB * SEQ * DIM * sizeof(float));

  const dim3 gemmBlk(256);
  const dim3 gGNN((NNODE + BM - 1) / BM, DIM / BN);   // 391 x 4
  const dim3 gSEQ((BB * SEQ + BM - 1) / BM, DIM / BN);

  // 1) h0 = relu(x @ Wp^T + bp)    [50000x768 @ 768x256]
  gemm_wmma_bf16<float><<<gGNN, gemmBlk, 0, stream>>>(x, Wp, bp, h0, NNODE, DIM, INF_, 1);

  float* hcur = h0;
  float* hnxt = h1;
  for (int l = 0; l < NL; ++l) {
    // h -> bf16 bits
    f32_to_bf16_kernel<<<4096, 256, 0, stream>>>((const float4*)hcur, (uint2*)hbf,
                                                 (size_t)NNODE * DIM / 4);
    // hs = h @ selfW[l]^T + selfb[l]
    gemm_wmma_bf16<unsigned short><<<gGNN, gemmBlk, 0, stream>>>(
        hbf, selfW + (size_t)l * DIM * DIM, selfb + (size_t)l * DIM, hs, NNODE, DIM, DIM, 0);
    // edge-type softmax weights
    softmax4_kernel<<<1, 32, 0, stream>>>(imp + l * NNET, ew);
    // agg = 0
    fillf_kernel<<<4096, 256, 0, stream>>>(agg, (size_t)NNODE * DIM, 0.f);
    for (int t = 0; t < NNET; ++t) {
      // msg = h @ edgeW[l,t]^T
      gemm_wmma_bf16<unsigned short><<<gGNN, gemmBlk, 0, stream>>>(
          hbf, edgeW + ((size_t)l * NNET + t) * DIM * DIM, nullptr, msg, NNODE, DIM, DIM, 0);
      const int* src = eidx + (size_t)t * 2 * NEDGE;
      const int* dst = src + NEDGE;
      scatter_edges_kernel<<<NEDGE, DIM, 0, stream>>>(msg, src, dst, ew + t, agg);
    }
    // combine + layernorm + residual
    combine_ln_kernel<<<NNODE, DIM, 0, stream>>>(agg, hs,
        hcur, gamma + (size_t)l * DIM, beta + (size_t)l * DIM, hnxt);
    float* tmp = hcur; hcur = hnxt; hnxt = tmp;
  }

  // 2) batch mean pooling -> gf [8x256]
  fillf_kernel<<<16, 256, 0, stream>>>(sums, (size_t)BB * DIM, 0.f);
  fillf_kernel<<<1, 32, 0, stream>>>(cnts, (size_t)BB, 0.f);
  pool_sum_kernel<<<NNODE, DIM, 0, stream>>>(hcur, batch, sums, cnts);
  pool_div_kernel<<<BB, DIM, 0, stream>>>(sums, cnts, gf);

  // 3) sequence branch: s0 = seqf @ Ws^T + bs  [1024x768 @ 768x256]
  gemm_wmma_bf16<float><<<gSEQ, gemmBlk, 0, stream>>>(seqf, Ws, bs, s0, BB * SEQ, DIM, INF_, 0);

  // 4) two BiLSTM layers (each layer: both dirs concurrently, grid=2)
  lstm_layer_kernel<<<2, 1024, 0, stream>>>(s0,
      Wih, Whh, bih, bhh, s1);
  lstm_layer_kernel<<<2, 1024, 0, stream>>>(s1,
      Wih + (size_t)1 * 2 * 4 * HH * (2 * HH),
      Whh + (size_t)1 * 2 * 4 * HH * HH,
      bih + 1 * 2 * 4 * HH,
      bhh + 1 * 2 * 4 * HH, s2);

  // 5) head: out[8x14]
  head_kernel<<<BB, DIM, 0, stream>>>(gf, s2, fW1, fb1, fW2, fb2, Wc, bc, out);
}
